// EnhancedFinGAT_18562848653562
// MI455X (gfx1250) — matrix operand order repaired and checked by reading, not compiled, not run on hardware
//
#include <hip/hip_runtime.h>
#include <hip/hip_bf16.h>

#define HID 128
#define HEADS 4
#define DH 32
#define LAYERS 4
#define NEG_SLOPE 0.2f

typedef __attribute__((ext_vector_type(16))) __bf16 v16bf;
typedef __attribute__((ext_vector_type(8)))  __bf16 v8bf;
typedef __attribute__((ext_vector_type(8)))  float  v8f;

// Monotonic float<->uint mapping so segment-max can use native u32 atomicMax.
__device__ __forceinline__ unsigned flip_f32(float f) {
  unsigned u = __float_as_uint(f);
  return u ^ ((u & 0x80000000u) ? 0xFFFFFFFFu : 0x80000000u);
}
__device__ __forceinline__ float unflip_f32(unsigned u) {
  unsigned v = (u & 0x80000000u) ? (u ^ 0x80000000u) : ~u;
  return __uint_as_float(v);
}

// A fragment loader: contiguous K order ->
// lanes 0-15 hold K = {kb..kb+15} with kb = ks*32;
// lanes 16-31 the same shifted by 16. Two 16B loads, no conversion.
__device__ __forceinline__ v16bf load_a_frag(const __bf16* arow, int ks, int half) {
  const int kb = ks * 32 + half * 16;
  v8bf lo = *(const v8bf*)(arow + kb);
  v8bf hi = *(const v8bf*)(arow + kb + 8);
  return __builtin_shufflevector(lo, hi, 0, 1, 2, 3, 4, 5, 6, 7,
                                 8, 9, 10, 11, 12, 13, 14, 15);
}

// Fused dual GEMM: C1 = A@W1, C2 = A@W2. A: [nrows][128] bf16, W*: [128][128] f32.
// 4 waves/block; each wave owns one 16x16 tile of both outputs (8 WMMA/wave).
// Both weight column-slices staged in LDS transposed [n][k] -> contiguous 32B B frags.
__global__ void __launch_bounds__(128)
gemm128x2_wmma_bf16(const __bf16* __restrict__ A, const float* __restrict__ W1,
                    const float* __restrict__ W2, float* __restrict__ C1,
                    float* __restrict__ C2, int nrows)
{
  __shared__ __align__(32) __bf16 Wlds[2][16 * 128];   // 8 KB
  const int colTile = blockIdx.x & 7;
  const int rowTile = (blockIdx.x >> 3) * 4 + (threadIdx.x >> 5);
  const int lane = threadIdx.x & 31;
  {
    const int k = threadIdx.x;
    const float* w1 = W1 + (size_t)k * HID + colTile * 16;
    const float* w2 = W2 + (size_t)k * HID + colTile * 16;
    #pragma unroll
    for (int n = 0; n < 16; ++n) {
      Wlds[0][n * 128 + k] = (__bf16)w1[n];
      Wlds[1][n * 128 + k] = (__bf16)w2[n];
    }
  }
  __syncthreads();
  const int ntiles = nrows >> 4;
  if (rowTile >= ntiles) return;  // uniform per wave -> EXEC all-ones for WMMA

  const int half = lane >> 4;
  const int mloc = lane & 15;
  const __bf16* arow = A + (size_t)(rowTile * 16 + mloc) * HID;

  v8f acc1 = {}, acc2 = {};
  #pragma unroll
  for (int ks = 0; ks < 4; ++ks) {
    v16bf a = load_a_frag(arow, ks, half);
    const int bofs = mloc * 128 + ks * 32 + half * 16;
    v16bf b1 = *(const v16bf*)&Wlds[0][bofs];
    v16bf b2 = *(const v16bf*)&Wlds[1][bofs];
    acc1 = __builtin_amdgcn_wmma_f32_16x16x32_bf16(false, a, false, b1,
                                                   (short)0, acc1, false, false);
    acc2 = __builtin_amdgcn_wmma_f32_16x16x32_bf16(false, a, false, b2,
                                                   (short)0, acc2, false, false);
  }
  const int n = colTile * 16 + mloc;
  #pragma unroll
  for (int r = 0; r < 8; ++r) {
    const size_t o = (size_t)(rowTile * 16 + half * 8 + r) * HID + n;
    C1[o] = acc1[r];
    C2[o] = acc2[r];
  }
}

// Single GEMM with bias + residual (res may alias C): C = A@W + bias + res.
__global__ void __launch_bounds__(128)
gemm128_wmma_bf16(const __bf16* __restrict__ A, const float* __restrict__ W,
                  const float* __restrict__ bias, const float* __restrict__ res,
                  float* __restrict__ C, int nrows)
{
  __shared__ __align__(32) __bf16 Wlds[16 * 128];   // 4 KB
  const int colTile = blockIdx.x & 7;
  const int rowTile = (blockIdx.x >> 3) * 4 + (threadIdx.x >> 5);
  const int lane = threadIdx.x & 31;
  {
    const int k = threadIdx.x;
    const float* wr = W + (size_t)k * HID + colTile * 16;
    #pragma unroll
    for (int n = 0; n < 16; ++n)
      Wlds[n * 128 + k] = (__bf16)wr[n];
  }
  __syncthreads();
  const int ntiles = nrows >> 4;
  if (rowTile >= ntiles) return;

  const int half = lane >> 4;
  const int mloc = lane & 15;
  const __bf16* arow = A + (size_t)(rowTile * 16 + mloc) * HID;

  v8f acc = {};
  #pragma unroll
  for (int ks = 0; ks < 4; ++ks) {
    v16bf a = load_a_frag(arow, ks, half);
    v16bf b = *(const v16bf*)&Wlds[mloc * 128 + ks * 32 + half * 16];
    acc = __builtin_amdgcn_wmma_f32_16x16x32_bf16(false, a, false, b,
                                                  (short)0, acc, false, false);
  }
  const int n = colTile * 16 + mloc;
  const float bn = bias ? bias[n] : 0.f;
  #pragma unroll
  for (int r = 0; r < 8; ++r) {
    const size_t o = (size_t)(rowTile * 16 + half * 8 + r) * HID + n;
    float v = acc[r] + bn;
    if (res) v += res[o];
    C[o] = v;
  }
}

// f32 -> bf16 bulk convert (4 elems/thread).
__global__ void __launch_bounds__(256)
cvt_f32_bf16(const float* __restrict__ in, __bf16* __restrict__ out, int total4)
{
  const int i = blockIdx.x * 256 + threadIdx.x;
  if (i >= total4) return;
  float4 v = *(const float4*)(in + (size_t)i * 4);
  __bf16* o = out + (size_t)i * 4;
  o[0] = (__bf16)v.x; o[1] = (__bf16)v.y; o[2] = (__bf16)v.z; o[3] = (__bf16)v.w;
}

// Pass 1: per-edge attention logits + segment max over dst.
// One wave per edge: lane -> (head = lane/8, 4 dims = (lane%8)*4).
__global__ void __launch_bounds__(256)
gat_edge_score(const float* __restrict__ xl, const float* __restrict__ xr,
               const int* __restrict__ ei, const float* __restrict__ att,
               float* __restrict__ score, unsigned* __restrict__ smax,
               int E, int Etot)
{
  const int e = (blockIdx.x * 256 + threadIdx.x) >> 5;
  if (e >= Etot) return;
  const int lane = threadIdx.x & 31;
  const int h = lane >> 3, sub = lane & 7;
  int src, dst;
  if (e < E) { src = ei[e]; dst = ei[E + e]; } else { src = dst = e - E; }
  const int co = h * DH + sub * 4;
  float4 vl = *(const float4*)(xl + (size_t)src * HID + co);
  float4 vr = *(const float4*)(xr + (size_t)dst * HID + co);
  float4 va = *(const float4*)(att + co);
  float t, s = 0.f;
  t = vl.x + vr.x; s += (t > 0.f ? t : NEG_SLOPE * t) * va.x;
  t = vl.y + vr.y; s += (t > 0.f ? t : NEG_SLOPE * t) * va.y;
  t = vl.z + vr.z; s += (t > 0.f ? t : NEG_SLOPE * t) * va.z;
  t = vl.w + vr.w; s += (t > 0.f ? t : NEG_SLOPE * t) * va.w;
  s += __shfl_xor(s, 1, 32);
  s += __shfl_xor(s, 2, 32);
  s += __shfl_xor(s, 4, 32);
  if (sub == 0) {
    score[(size_t)e * HEADS + h] = s;
    atomicMax(smax + (size_t)dst * HEADS + h, flip_f32(s));
  }
}

// Pass 2: exp(score - max[dst]) in place + denom accumulation.
__global__ void __launch_bounds__(256)
gat_edge_exp(const int* __restrict__ ei, float* __restrict__ sc,
             const unsigned* __restrict__ smax, float* __restrict__ denom,
             int E, int Etot)
{
  const int idx = blockIdx.x * 256 + threadIdx.x;
  if (idx >= Etot * HEADS) return;
  const int e = idx >> 2, h = idx & 3;
  const int dst = (e < E) ? ei[E + e] : e - E;
  const float m = unflip_f32(smax[(size_t)dst * HEADS + h]);
  const float ex = __expf(sc[idx] - m);
  sc[idx] = ex;
  atomicAdd(denom + (size_t)dst * HEADS + h, ex);
}

// Pass 3: h[dst] += alpha * xl[src]; one wave per edge, 4 floats per lane.
__global__ void __launch_bounds__(256)
gat_edge_aggregate(const float* __restrict__ xl, const int* __restrict__ ei,
                   const float* __restrict__ expv, const float* __restrict__ denom,
                   float* __restrict__ hout, int E, int Etot)
{
  const int e = (blockIdx.x * 256 + threadIdx.x) >> 5;
  if (e >= Etot) return;
  const int lane = threadIdx.x & 31;
  const int h = lane >> 3, sub = lane & 7;
  int src, dst;
  if (e < E) { src = ei[e]; dst = ei[E + e]; } else { src = dst = e - E; }
  const float alpha = expv[(size_t)e * HEADS + h] / denom[(size_t)dst * HEADS + h];
  const int co = h * DH + sub * 4;
  float4 vl = *(const float4*)(xl + (size_t)src * HID + co);
  float* o = hout + (size_t)dst * HID + co;
  atomicAdd(o + 0, alpha * vl.x);
  atomicAdd(o + 1, alpha * vl.y);
  atomicAdd(o + 2, alpha * vl.z);
  atomicAdd(o + 3, alpha * vl.w);
}

__global__ void __launch_bounds__(256)
init_bias(float* __restrict__ h, const float* __restrict__ cb, int total)
{
  const int idx = blockIdx.x * 256 + threadIdx.x;
  if (idx < total) h[idx] = cb[idx & (HID - 1)];
}

// GraphNorm channel stats: thread = channel (coalesced), grid-stride over nodes.
__global__ void __launch_bounds__(128)
gn_stats(const float* __restrict__ h, float* __restrict__ sums,
         float* __restrict__ sumsq, int N)
{
  const int c = threadIdx.x;
  float s = 0.f, ss = 0.f;
  for (int n = blockIdx.x; n < N; n += gridDim.x) {
    const float v = h[(size_t)n * HID + c];
    s += v; ss += v * v;
  }
  atomicAdd(&sums[c], s);
  atomicAdd(&sumsq[c], ss);
}

// var(h - m*mean) = E[h^2] - (2m - m^2) * mean^2; fold affine into scale/shift.
__global__ void gn_finalize(const float* __restrict__ sums, const float* __restrict__ sumsq,
                            const float* __restrict__ w, const float* __restrict__ b,
                            const float* __restrict__ ms, float* __restrict__ scale,
                            float* __restrict__ shift, int N)
{
  const int c = threadIdx.x;
  const float inv = 1.f / (float)N;
  const float mean = sums[c] * inv;
  const float eh2 = sumsq[c] * inv;
  const float m = ms[c];
  const float var = eh2 - (2.f * m - m * m) * mean * mean;
  const float sc = w[c] * rsqrtf(var + 1e-5f);
  scale[c] = sc;
  shift[c] = b[c] - sc * m * mean;
}

// Fused GraphNorm affine + ELU, emitting bf16 directly for the skip GEMM.
__global__ void __launch_bounds__(256)
gn_elu_bf16(const float* __restrict__ h, const float* __restrict__ scale,
            const float* __restrict__ shift, __bf16* __restrict__ out, int total)
{
  const int idx = blockIdx.x * 256 + threadIdx.x;
  if (idx >= total) return;
  const int c = idx & (HID - 1);
  float y = h[idx] * scale[c] + shift[c];
  y = (y > 0.f) ? y : (__expf(y) - 1.f);
  out[idx] = (__bf16)y;
}

// Final fc: one wave per node, 128-dot via shuffle reduction.
__global__ void __launch_bounds__(256)
fc_head(const float* __restrict__ x, const float* __restrict__ W,
        const float* __restrict__ b, float* __restrict__ out, int N)
{
  const int node = (blockIdx.x * 256 + threadIdx.x) >> 5;
  if (node >= N) return;
  const int lane = threadIdx.x & 31;
  float4 xv = *(const float4*)(x + (size_t)node * HID + lane * 4);
  float4 wv = *(const float4*)(W + lane * 4);
  float s = xv.x * wv.x + xv.y * wv.y + xv.z * wv.z + xv.w * wv.w;
  #pragma unroll
  for (int m = 1; m < 32; m <<= 1) s += __shfl_xor(s, m, 32);
  if (lane == 0) out[node] = s + b[0];
}

extern "C" void kernel_launch(void* const* d_in, const int* in_sizes, int n_in,
                              void* d_out, int out_size, void* d_ws, size_t ws_size,
                              hipStream_t stream) {
  (void)n_in; (void)out_size; (void)ws_size;
  const float* x_in  = (const float*)d_in[0];
  const int*   ei    = (const int*)d_in[1];
  const float* Wl    = (const float*)d_in[2];
  const float* Wr    = (const float*)d_in[3];
  const float* att   = (const float*)d_in[4];
  const float* convb = (const float*)d_in[5];
  const float* gnw   = (const float*)d_in[6];
  const float* gnb   = (const float*)d_in[7];
  const float* gnms  = (const float*)d_in[8];
  const float* skW   = (const float*)d_in[9];
  const float* skb   = (const float*)d_in[10];
  const float* fcW   = (const float*)d_in[11];
  const float* fcb   = (const float*)d_in[12];

  const int N = in_sizes[0] / HID;
  const int E = in_sizes[1] / 2;
  const int Etot = E + N;

  // Workspace: x | xl | xr-then-h | xbf(bf16, also elu-out) | score | smax | denom | stats
  char* p = (char*)d_ws;
  auto carve = [&](size_t bytes) { void* r = (void*)p; p += (bytes + 255) & ~(size_t)255; return r; };
  float*    x     = (float*)carve((size_t)N * HID * 4);
  float*    xl    = (float*)carve((size_t)N * HID * 4);
  float*    xrh   = (float*)carve((size_t)N * HID * 4);
  __bf16*   xbf   = (__bf16*)carve((size_t)N * HID * 2);
  float*    sc    = (float*)carve((size_t)Etot * HEADS * 4);
  unsigned* smax  = (unsigned*)carve((size_t)N * HEADS * 4);
  float*    denom = (float*)carve((size_t)N * HEADS * 4);
  float*    sums  = (float*)carve(HID * 4);
  float*    sumsq = (float*)carve(HID * 4);
  float*    scale = (float*)carve(HID * 4);
  float*    shift = (float*)carve(HID * 4);

  hipMemcpyAsync(x, x_in, (size_t)N * HID * 4, hipMemcpyDeviceToDevice, stream);

  const int rowTiles = N / 16;
  const int gemmGrid = ((rowTiles + 3) / 4) * 8;
  const int edgeWaveBlocks = (int)(((long long)Etot * 32 + 255) / 256);
  const int ehBlocks = (Etot * HEADS + 255) / 256;
  const int elemBlocks = (N * HID + 255) / 256;
  const int cvtBlocks = (N * HID / 4 + 255) / 256;

  for (int l = 0; l < LAYERS; ++l) {
    const float* Wl_l = Wl + (size_t)l * HID * HID;
    const float* Wr_l = Wr + (size_t)l * HID * HID;
    // x -> bf16 once; fused xl = x@Wl, xr = x@Wr (shared A fragments, 8 WMMA/wave)
    cvt_f32_bf16<<<cvtBlocks, 256, 0, stream>>>(x, xbf, N * HID / 4);
    gemm128x2_wmma_bf16<<<gemmGrid, 128, 0, stream>>>(xbf, Wl_l, Wr_l, xl, xrh, N);
    // attention: score + segment max (flipped-uint atomicMax; 0 == -inf)
    hipMemsetAsync(smax, 0, (size_t)N * HEADS * 4, stream);
    gat_edge_score<<<edgeWaveBlocks, 256, 0, stream>>>(
        xl, xrh, ei, att + (size_t)l * HID, sc, smax, E, Etot);
    // xr no longer needed -> reuse buffer as h, seeded with conv bias
    init_bias<<<elemBlocks, 256, 0, stream>>>(xrh, convb + (size_t)l * HID, N * HID);
    hipMemsetAsync(denom, 0, (size_t)N * HEADS * 4, stream);
    gat_edge_exp<<<ehBlocks, 256, 0, stream>>>(ei, sc, smax, denom, E, Etot);
    gat_edge_aggregate<<<edgeWaveBlocks, 256, 0, stream>>>(xl, ei, sc, denom, xrh, E, Etot);
    // GraphNorm + ELU -> bf16 (feeds skip GEMM directly)
    hipMemsetAsync(sums, 0, HID * 4, stream);
    hipMemsetAsync(sumsq, 0, HID * 4, stream);
    gn_stats<<<256, 128, 0, stream>>>(xrh, sums, sumsq, N);
    gn_finalize<<<1, 128, 0, stream>>>(sums, sumsq, gnw + (size_t)l * HID,
                                       gnb + (size_t)l * HID, gnms + (size_t)l * HID,
                                       scale, shift, N);
    gn_elu_bf16<<<elemBlocks, 256, 0, stream>>>(xrh, scale, shift, xbf, N * HID);
    // skip: x = x + elu(h) @ skip_W + skip_b  (in-place residual, WMMA)
    gemm128_wmma_bf16<<<gemmGrid, 128, 0, stream>>>(xbf, skW, skb, x, x, N);
  }
  fc_head<<<(N * 32 + 255) / 256, 256, 0, stream>>>(x, fcW, fcb, (float*)d_out, N);
}
